// MultiHeadAttention_53042846105789
// MI455X (gfx1250) — matrix operand order repaired and checked by reading, not compiled, not run on hardware
//
#include <hip/hip_runtime.h>
#include <hip/hip_bf16.h>

// ---------------------------------------------------------------------------
// MHA for MI455X (gfx1250): all matmuls on v_wmma_f32_16x16x32_bf16 (wave32).
// B=2, S=2048, D=1024, H=16, DH=64.
// ---------------------------------------------------------------------------

typedef __attribute__((ext_vector_type(16))) __bf16 v16bf;
typedef __attribute__((ext_vector_type(8)))  __bf16 v8bf;
typedef __attribute__((ext_vector_type(4)))  __bf16 v4bf;
typedef __attribute__((ext_vector_type(8)))  float  v8f;
typedef __attribute__((ext_vector_type(4)))  int    v4i;

#define Bn   2
#define Sn   2048
#define Dn   1024
#define Hn   16
#define DHn  64
#define BHn  (Bn * Hn)
#define TK   64                 // staged k-depth in the GEMM

// output modes for the generic GEMM
#define OUT_SPLIT_HEADS    0    // bf16 out[((b*H+h)*S+s)*DH+dh]   (Q, K)
#define OUT_SPLIT_HEADS_T  1    // bf16 out[((b*H+h)*DH+dh)*S+s]   (V transposed)
#define OUT_DENSE_F32      2    // f32  out[m*N+n]                 (final proj)

#if defined(__has_builtin)
# if __has_builtin(__builtin_amdgcn_global_load_async_to_lds_b128)
#  define HAVE_ASYNC_LDS 1
# endif
#endif

static __device__ __forceinline__ v16bf frag_cat(v8bf lo, v8bf hi) {
    return __builtin_shufflevector(lo, hi, 0, 1, 2, 3, 4, 5, 6, 7,
                                           8, 9, 10, 11, 12, 13, 14, 15);
}

static __device__ __forceinline__ v8f wmma_bf16(v16bf a, v16bf b, v8f c) {
    // (neg_a, A, neg_b, B, c_mod, C, reuse_a, reuse_b)
    return __builtin_amdgcn_wmma_f32_16x16x32_bf16(false, a, false, b,
                                                   (short)0, c, false, false);
}

// A fragment (16x32, lane holds row lane&15; K = {0..7,16..23} / {8..15,24..31})
static __device__ __forceinline__ v16bf load_afrag(const __bf16* rowbase, int lane) {
    const int kb = (lane & 16) ? 8 : 0;
    v8bf lo = *(const v8bf*)(rowbase + kb);            // 16B aligned
    v8bf hi = *(const v8bf*)(rowbase + kb + 16);
    return frag_cat(lo, hi);
}

// B fragment (32x16, lane holds column lane&15; K = {0..7,16..23} / {8..15,24..31})
static __device__ __forceinline__ v16bf load_bfrag(const __bf16* colbase, int lane) {
    const int kb = (lane & 16) ? 8 : 0;
    v8bf lo = *(const v8bf*)(colbase + kb);            // 16B aligned
    v8bf hi = *(const v8bf*)(colbase + kb + 16);
    return frag_cat(lo, hi);
}

static __device__ __forceinline__ v4bf cvt4(float a, float b, float c, float d) {
    v4bf o;
    o[0] = (__bf16)a; o[1] = (__bf16)b; o[2] = (__bf16)c; o[3] = (__bf16)d;
    return o;
}

// ---------------------------------------------------------------------------
// Generic GEMM:  C[M,N] = A[M,K](f32->bf16) * B[K,N](f32->bf16) + bias
// block = 128 threads (4 waves), block tile 64x64, staged k-depth 64.
// Register double-buffered: next k-tile's global loads are issued before the
// WMMA chain of the current tile, hiding HBM/L2 latency behind matrix math.
// ---------------------------------------------------------------------------
__launch_bounds__(128)
__global__ void gemm_bf16_wmma(const float* __restrict__ A,
                               const float* __restrict__ B,
                               const float* __restrict__ bias,
                               __bf16* __restrict__ outb,
                               float*  __restrict__ outf,
                               int M, int N, int K, int mode)
{
    __shared__ __align__(32) __bf16 As[64][TK];   // A tile, row-major
    __shared__ __align__(32) __bf16 Bt[64][TK];   // B tile transposed: Bt[n][k]

    const int tid   = threadIdx.x;
    const int lane  = tid & 31;
    const int w     = tid >> 5;            // 0..3
    const int rlane = lane & 15;
    const int m0    = blockIdx.y * 64;
    const int n0    = blockIdx.x * 64;

    v8f acc[4] = {};

    float4 ra[8];          // staged A values (row-contiguous float4)
    float  rb[8][4];       // staged B values (k-contiguous, n-coalesced loads)

    auto load_tiles = [&](int k0) {
#pragma unroll
        for (int it = 0; it < 8; ++it) {
            const int i  = tid + it * 128;         // 1024 float4 groups
            const int r  = i >> 4;                 // 16 groups per row
            const int c4 = (i & 15) * 4;
            ra[it] = *(const float4*)&A[(size_t)(m0 + r) * K + k0 + c4];
        }
#pragma unroll
        for (int it = 0; it < 8; ++it) {
            const int i  = tid + it * 128;         // 1024 (n, k4) groups
            const int n  = i & 63;                 // consecutive tids -> consecutive n
            const int k4 = (i >> 6) * 4;
            const size_t base = (size_t)(k0 + k4) * N + (n0 + n);
            rb[it][0] = B[base];
            rb[it][1] = B[base + N];
            rb[it][2] = B[base + 2 * (size_t)N];
            rb[it][3] = B[base + 3 * (size_t)N];
        }
    };

    load_tiles(0);   // prologue

    for (int k0 = 0; k0 < K; k0 += TK) {
        // ---- dump staged registers into LDS (fp32 -> packed bf16) ----
#pragma unroll
        for (int it = 0; it < 8; ++it) {
            const int i  = tid + it * 128;
            const int r  = i >> 4;
            const int c4 = (i & 15) * 4;
            *(v4bf*)&As[r][c4] = cvt4(ra[it].x, ra[it].y, ra[it].z, ra[it].w);
        }
#pragma unroll
        for (int it = 0; it < 8; ++it) {
            const int i  = tid + it * 128;
            const int n  = i & 63;
            const int k4 = (i >> 6) * 4;
            *(v4bf*)&Bt[n][k4] = cvt4(rb[it][0], rb[it][1], rb[it][2], rb[it][3]);
        }
        __syncthreads();

        // ---- issue next tile's global loads; they overlap the WMMAs below ----
        if (k0 + TK < K) load_tiles(k0 + TK);

        const v16bf a0 = load_afrag(&As[w * 16 + rlane][0], lane);
        const v16bf a1 = load_afrag(&As[w * 16 + rlane][32], lane);
#pragma unroll
        for (int nt = 0; nt < 4; ++nt) {
            const v16bf b0 = load_bfrag(&Bt[nt * 16 + rlane][0], lane);
            const v16bf b1 = load_bfrag(&Bt[nt * 16 + rlane][32], lane);
            acc[nt] = wmma_bf16(a0, b0, acc[nt]);
            acc[nt] = wmma_bf16(a1, b1, acc[nt]);
        }
        __syncthreads();
    }

    // writeback: C layout — element e: row = e + 8*(lane>=16), col = lane&15
#pragma unroll
    for (int nt = 0; nt < 4; ++nt) {
#pragma unroll
        for (int e = 0; e < 8; ++e) {
            const int row = ((lane & 16) ? 8 : 0) + e;
            const int m   = m0 + w * 16 + row;
            const int n   = n0 + nt * 16 + rlane;
            const float val = acc[nt][e] + bias[n];
            if (mode == OUT_DENSE_F32) {
                outf[(size_t)m * N + n] = val;
            } else {
                const int b  = m >> 11;          // m / S
                const int s  = m & (Sn - 1);
                const int h  = n >> 6;           // n / DH
                const int dh = n & (DHn - 1);
                size_t idx;
                if (mode == OUT_SPLIT_HEADS)
                    idx = (((size_t)(b * Hn + h) * Sn) + s) * DHn + dh;
                else
                    idx = (((size_t)(b * Hn + h) * DHn) + dh) * Sn + s;
                outb[idx] = (__bf16)val;
            }
        }
    }
}

// ---------------------------------------------------------------------------
// Attention: one block per (bh, 16-row q tile); 256 threads = 8 waves.
// scores kept fp32 in LDS (16 x 2048 = 128 KB); dynamic shared (320 KB WGP LDS).
// ---------------------------------------------------------------------------
__launch_bounds__(256)
__global__ void attention_wmma(const __bf16* __restrict__ Qh,   // [BH][S][DH]
                               const __bf16* __restrict__ Kh,   // [BH][S][DH]
                               const __bf16* __restrict__ Vt,   // [BH][DH][S]
                               const float*  __restrict__ mask, // [B][1][S][S]
                               float* __restrict__ attnO)       // [B*S][D]
{
    extern __shared__ __align__(32) char smem[];
    float*  scores = (float*)smem;                                   // 16*2048 f32
    __bf16* Qs     = (__bf16*)(smem + (size_t)16 * Sn * 4);          // 16*64 bf16
    float*  Os     = (float*)(smem + (size_t)16 * Sn * 4 + 16 * 64 * 2); // 16*64 f32
    float*  red    = Os + 16 * 64;                                   // 16*16
    float*  rowmax = red + 256;                                      // 16
    float*  rowsum = rowmax + 16;                                    // 16

    const int tid   = threadIdx.x;
    const int lane  = tid & 31;
    const int w     = tid >> 5;            // 0..7
    const int rlane = lane & 15;
    const int kb    = (lane & 16) ? 8 : 0;
    const int q0    = blockIdx.x * 16;
    const int bh    = blockIdx.y;
    const int b     = bh / Hn;
    const int h     = bh % Hn;

    const size_t qkBase = (size_t)bh * Sn * DHn;
    const size_t vBase  = (size_t)bh * DHn * Sn;

    // ---- stage Q tile (16x64 bf16 = 2 KiB, contiguous) into LDS ----
#if defined(HAVE_ASYNC_LDS)
    if (tid < 128) {   // 128 lanes x 16 B  (ASYNCcnt-tracked DMA to LDS)
        const __bf16* gq = Qh + qkBase + (size_t)q0 * DHn + (size_t)tid * 8;
        __builtin_amdgcn_global_load_async_to_lds_b128(
            (__attribute__((address_space(1))) v4i*)(void*)gq,
            (__attribute__((address_space(3))) v4i*)(void*)(Qs + tid * 8),
            0, 0);
    }
#else
    for (int i = tid; i < 16 * DHn; i += 256)
        Qs[i] = Qh[qkBase + (size_t)q0 * DHn + i];
#endif
    // zero output accumulator tile while the copy is in flight
    for (int i = tid; i < 16 * DHn; i += 256) Os[i] = 0.0f;
#if defined(HAVE_ASYNC_LDS)
# if __has_builtin(__builtin_amdgcn_s_wait_asynccnt)
    __builtin_amdgcn_s_wait_asynccnt(0);
# else
    asm volatile("s_wait_asynccnt 0x0" ::: "memory");
# endif
#endif
    __syncthreads();

    // Q fragments for k = 0..31 and 32..63 (shared row tile per wave)
    const v16bf aq0 = load_afrag(&Qs[rlane * DHn], lane);
    const v16bf aq1 = load_afrag(&Qs[rlane * DHn + 32], lane);

    // ---- Phase 1: logits = scale * Q K^T + mask*(-1e9) -> LDS (fp32) ----
    const float scale = 0.125f;  // 1/sqrt(64)
#pragma unroll 2
    for (int ct = w; ct < Sn / 16; ct += 8) {
        const int sk0 = ct * 16;
        const __bf16* kp = &Kh[qkBase + (size_t)(sk0 + rlane) * DHn];
        const v16bf b0 = load_bfrag(kp, lane);        // K rows 0..31 of DH
        const v16bf b1 = load_bfrag(kp + 32, lane);   // K rows 32..63
        v8f acc = {};
        acc = wmma_bf16(aq0, b0, acc);
        acc = wmma_bf16(aq1, b1, acc);
#pragma unroll
        for (int e = 0; e < 8; ++e) {
            const int row = ((lane & 16) ? 8 : 0) + e;
            const int col = sk0 + rlane;
            const float mv = mask[((size_t)b * Sn + (q0 + row)) * Sn + col];
            scores[row * Sn + col] = acc[e] * scale - 1.0e9f * mv;
        }
    }
    __syncthreads();

    // ---- Phase 2: row softmax (16 rows x 16 threads each) ----
    {
        const int r = tid >> 4, c = tid & 15;
        float mx = -3.4e38f;
        for (int j = c; j < Sn; j += 16) mx = fmaxf(mx, scores[r * Sn + j]);
        red[r * 16 + c] = mx;
        __syncthreads();
        if (c == 0) {
            float m2 = red[r * 16];
            for (int j = 1; j < 16; ++j) m2 = fmaxf(m2, red[r * 16 + j]);
            rowmax[r] = m2;
        }
        __syncthreads();
        const float rm = rowmax[r];
        float sum = 0.0f;
        for (int j = c; j < Sn; j += 16) {
            const float p = __expf(scores[r * Sn + j] - rm);
            scores[r * Sn + j] = p;
            sum += p;
        }
        red[r * 16 + c] = sum;
        __syncthreads();
        if (c == 0) {
            float s2 = 0.0f;
            for (int j = 0; j < 16; ++j) s2 += red[r * 16 + j];
            rowsum[r] = s2;
        }
        __syncthreads();
        const float inv = 1.0f / rowsum[r];
        for (int j = c; j < Sn; j += 16) scores[r * Sn + j] *= inv;
        __syncthreads();
    }

    // ---- Phase 3: out = P (16 x S) * V (S x 64), split-K across waves ----
    v8f oacc[4] = {};
#pragma unroll 2
    for (int kt = w; kt < Sn / 32; kt += 8) {
        const int kk = kt * 32;
        // A fragment from fp32 scores (convert to bf16 per ISA A layout)
        v16bf ap;
        {
            union { v16bf v; __bf16 e[16]; } u;
            const float* sp = &scores[rlane * Sn + kk + kb];
#pragma unroll
            for (int j = 0; j < 8; ++j) {
                u.e[j]     = (__bf16)sp[j];
                u.e[8 + j] = (__bf16)sp[16 + j];
            }
            ap = u.v;
        }
#pragma unroll
        for (int nt = 0; nt < 4; ++nt) {
            const __bf16* vp = &Vt[vBase + (size_t)(nt * 16 + rlane) * Sn + kk];
            const v16bf bf = load_bfrag(vp, lane);
            oacc[nt] = wmma_bf16(ap, bf, oacc[nt]);
        }
    }
    // cross-wave reduce via LDS float atomics (ds_add_f32)
#pragma unroll
    for (int nt = 0; nt < 4; ++nt) {
#pragma unroll
        for (int e = 0; e < 8; ++e) {
            const int row = ((lane & 16) ? 8 : 0) + e;
            const int dh  = nt * 16 + rlane;
            atomicAdd(&Os[row * DHn + dh], oacc[nt][e]);
        }
    }
    __syncthreads();

    // merged-head fp32 output for the final projection GEMM
    for (int i = tid; i < 16 * DHn; i += 256) {
        const int row = i >> 6, dh = i & 63;
        attnO[((size_t)(b * Sn + q0 + row)) * Dn + h * DHn + dh] = Os[i];
    }
}

// ---------------------------------------------------------------------------
extern "C" void kernel_launch(void* const* d_in, const int* in_sizes, int n_in,
                              void* d_out, int out_size, void* d_ws, size_t ws_size,
                              hipStream_t stream)
{
    const float* q    = (const float*)d_in[0];
    const float* k    = (const float*)d_in[1];
    const float* v    = (const float*)d_in[2];
    const float* mask = (const float*)d_in[3];
    const float* Wq   = (const float*)d_in[4];
    const float* bq   = (const float*)d_in[5];
    const float* Wk   = (const float*)d_in[6];
    const float* bk   = (const float*)d_in[7];
    const float* Wv   = (const float*)d_in[8];
    const float* bv   = (const float*)d_in[9];
    const float* Wo   = (const float*)d_in[10];
    const float* bo   = (const float*)d_in[11];

    // workspace carve: Qh/Kh/Vt bf16 (8 MiB each) + attnO fp32 (16 MiB)
    char* ws = (char*)d_ws;
    __bf16* Qh    = (__bf16*)(ws);
    __bf16* Kh    = (__bf16*)(ws + ((size_t)8  << 20));
    __bf16* Vt    = (__bf16*)(ws + ((size_t)16 << 20));
    float*  attnO = (float*) (ws + ((size_t)24 << 20));

    const int M = Bn * Sn, N = Dn, K = Dn;
    dim3 gg(N / 64, M / 64), gb(128);

    gemm_bf16_wmma<<<gg, gb, 0, stream>>>(q, Wq, bq, Qh, nullptr, M, N, K, OUT_SPLIT_HEADS);
    gemm_bf16_wmma<<<gg, gb, 0, stream>>>(k, Wk, bk, Kh, nullptr, M, N, K, OUT_SPLIT_HEADS);
    gemm_bf16_wmma<<<gg, gb, 0, stream>>>(v, Wv, bv, Vt, nullptr, M, N, K, OUT_SPLIT_HEADS_T);

    const size_t smem = (size_t)16 * Sn * sizeof(float)      // scores
                      + (size_t)16 * DHn * sizeof(__bf16)    // Qs
                      + (size_t)16 * DHn * sizeof(float)     // Os
                      + (256 + 32) * sizeof(float);          // red + rowmax/rowsum
    (void)hipFuncSetAttribute((const void*)attention_wmma,
                              hipFuncAttributeMaxDynamicSharedMemorySize, (int)smem);
    attention_wmma<<<dim3(Sn / 16, BHn), 256, smem, stream>>>(Qh, Kh, Vt, mask, attnO);

    gemm_bf16_wmma<<<gg, gb, 0, stream>>>(attnO, Wo, bo, nullptr, (float*)d_out,
                                          M, N, K, OUT_DENSE_F32);
}
